// Net_40561671143795
// MI455X (gfx1250) — matrix operand order repaired and checked by reading, not compiled, not run on hardware
//
#include <hip/hip_runtime.h>
#include <hip/hip_bf16.h>

// CDNA5 / gfx1250 LSTM: one wave32 per 16 batch rows, WMMA f16->f32 for both
// the input projection (xe@Wx) and the recurrent matmul (h@Wh), chained C.
// Gate nonlinearities use gfx1250's hardware v_tanh_f32.

typedef _Float16 v16h __attribute__((ext_vector_type(16)));
typedef _Float16 v8h  __attribute__((ext_vector_type(8)));
typedef float    v8f  __attribute__((ext_vector_type(8)));
typedef int      v4i  __attribute__((ext_vector_type(4)));

#define T_LEN 512
#define EMBD  32
#define HIDD  32
#define VOCAB 101
#define FOURH 128

__device__ __forceinline__ float fast_tanh(float x) { return __builtin_amdgcn_tanhf(x); }

// sigmoid(g + b) = 0.5 + 0.5*tanh(0.5*g + 0.5*b); ah = 0.5*b precomputed
__device__ __forceinline__ float sig_b(float g, float ah) {
    return __builtin_fmaf(0.5f, fast_tanh(__builtin_fmaf(0.5f, g, ah)), 0.5f);
}
// tanh(g + b); ah = b
__device__ __forceinline__ float tanh_b(float g, float ah) {
    return fast_tanh(g + ah);
}

__global__ __launch_bounds__(32) void lstm_wmma_kernel(
    const int*   __restrict__ x,     // [B, T]
    const float* __restrict__ emb,   // [VOCAB, EMB]
    const float* __restrict__ Wx,    // [EMB, 4H]
    const float* __restrict__ Wh,    // [HID, 4H]
    const float* __restrict__ bg,    // [4H]
    const float* __restrict__ W_fc,  // [HID, 2]
    const float* __restrict__ b_fc,  // [2]
    float*       __restrict__ out)   // [B, 2]
{
    __shared__ __align__(16) int      ids[16 * T_LEN];        // 32 KB: token ids for this tile
    __shared__ __align__(16) _Float16 embh[VOCAB * EMBD];     // 6.3 KB: f16 embedding table
    __shared__ __align__(16) _Float16 hbuf[16 * HIDD];        // 1 KB: h layout bounce buffer

    const int lane = threadIdx.x;       // 0..31
    const int b0   = blockIdx.x * 16;   // batch tile base
    const int mrow = lane & 15;         // 0..15
    const int hi   = lane >> 4;         // 0 or 1 (lane half)

    // ---- stage token ids: 16 contiguous rows of x -> LDS (b128 copies) ----
    {
        const v4i* src = (const v4i*)(x + (long)b0 * T_LEN);
        v4i* dst = (v4i*)ids;
        #pragma unroll 4
        for (int i = lane; i < (16 * T_LEN) / 4; i += 32) dst[i] = src[i];
    }
    // ---- stage embedding table as f16 (101*32 = 3232, uniform trip count) ----
    for (int i = lane; i < VOCAB * EMBD; i += 32) embh[i] = (_Float16)emb[i];

    // ---- build loop-invariant B fragments for Wx, Wh and folded bias ----
    // B layout (32x16 f16, 8 VGPRs): lanes 0-15 K=0..15, lanes 16-31 K=16..31,
    // VGPR r holds K = kbase+2r (low half), kbase+2r+1 (high half), col N = lane&15.
    v16h wx[8], wh[8];
    float abias[8];
    {
        const int kbase = hi * 16;
        #pragma unroll
        for (int n = 0; n < 8; ++n) {
            const int c_ = n * 16 + mrow;
            #pragma unroll
            for (int r = 0; r < 8; ++r) {
                const int k0 = kbase + 2 * r;
                wx[n][2*r]   = (_Float16)Wx[k0 * FOURH + c_];
                wx[n][2*r+1] = (_Float16)Wx[(k0 + 1) * FOURH + c_];
                wh[n][2*r]   = (_Float16)Wh[k0 * FOURH + c_];
                wh[n][2*r+1] = (_Float16)Wh[(k0 + 1) * FOURH + c_];
            }
            // tiles 0-3: i,f (sigmoid, fold 0.5*b); 4-5: g (tanh, fold b);
            // 6-7: o (sigmoid, fold 0.5*b)
            const float kk = (n == 4 || n == 5) ? 1.0f : 0.5f;
            abias[n] = kk * bg[c_];
        }
    }

    v8f  ct0 = {}, ct1 = {};   // cell state tiles (f32, C/D layout), cols 0-15 / 16-31
    v16h hF  = {};             // h as 16x32 f16 A fragment (h0 = 0)
    v8f  zeroC = {};

    #pragma unroll 1
    for (int t = 0; t < T_LEN; ++t) {
        // ---- gather embedding row -> A fragment (16x32 f16) ----
        // A layout: lane holds row M=lane&15; V0-3 <- K 0..7 (lanes<16) / 16..23,
        // V4-7 <- K 8..15 / 24..31 : two 16B LDS loads.
        const int id = ids[mrow * T_LEN + t];
        const v8h* ep = (const v8h*)(embh + id * EMBD + hi * 16);
        const v8h elo = ep[0];
        const v8h ehi = ep[1];          // +8 halves
        v16h aF;
        #pragma unroll
        for (int i = 0; i < 8; ++i) { aF[i] = elo[i]; aF[8 + i] = ehi[i]; }

        // ---- g = xe@Wx + h@Wh  (bias folded into nonlinearity) ----
        v8f g[8];
        #pragma unroll
        for (int n = 0; n < 8; ++n) {
            v8f acc = __builtin_amdgcn_wmma_f32_16x16x32_f16(
                false, aF, false, wx[n], (short)0, zeroC, false, false);
            g[n] = __builtin_amdgcn_wmma_f32_16x16x32_f16(
                false, hF, false, wh[n], (short)0, acc, false, false);
        }

        // ---- gates: c' = sig(f)*c + sig(i)*tanh(g); h = sig(o)*tanh(c') ----
        v8f hn0, hn1;
        #pragma unroll
        for (int r = 0; r < 8; ++r) {
            {
                const float si = sig_b (g[0][r], abias[0]);
                const float sf = sig_b (g[2][r], abias[2]);
                const float tg = tanh_b(g[4][r], abias[4]);
                const float so = sig_b (g[6][r], abias[6]);
                const float cn = __builtin_fmaf(sf, ct0[r], si * tg);
                ct0[r] = cn;
                hn0[r] = so * fast_tanh(cn);
            }
            {
                const float si = sig_b (g[1][r], abias[1]);
                const float sf = sig_b (g[3][r], abias[3]);
                const float tg = tanh_b(g[5][r], abias[5]);
                const float so = sig_b (g[7][r], abias[7]);
                const float cn = __builtin_fmaf(sf, ct1[r], si * tg);
                ct1[r] = cn;
                hn1[r] = so * fast_tanh(cn);
            }
        }

        // ---- C/D layout -> row-major f16 h in LDS (same-wave LDS is in-order) ----
        #pragma unroll
        for (int r = 0; r < 8; ++r) {
            const int row = r + hi * 8;
            hbuf[row * HIDD + mrow]      = (_Float16)hn0[r];
            hbuf[row * HIDD + mrow + 16] = (_Float16)hn1[r];
        }
        // ---- reload h as next step's A fragment ----
        const v8h* hp = (const v8h*)(hbuf + mrow * HIDD + hi * 16);
        const v8h hlo = hp[0];
        const v8h hhi = hp[1];
        #pragma unroll
        for (int i = 0; i < 8; ++i) { hF[i] = hlo[i]; hF[8 + i] = hhi[i]; }
    }

    // ---- final FC: out[b0+row][j] = h[row,:]·W_fc[:,j] + b_fc[j], j = lane half ----
    float acc = b_fc[hi];
    #pragma unroll
    for (int k = 0; k < HIDD; ++k)
        acc = __builtin_fmaf((float)hbuf[mrow * HIDD + k], W_fc[k * 2 + hi], acc);
    out[(b0 + mrow) * 2 + hi] = acc;
}

extern "C" void kernel_launch(void* const* d_in, const int* in_sizes, int n_in,
                              void* d_out, int out_size, void* d_ws, size_t ws_size,
                              hipStream_t stream) {
    const int*   x    = (const int*)  d_in[0];
    const float* emb  = (const float*)d_in[1];
    const float* Wx   = (const float*)d_in[2];
    const float* Wh   = (const float*)d_in[3];
    const float* bg   = (const float*)d_in[4];
    const float* W_fc = (const float*)d_in[5];
    const float* b_fc = (const float*)d_in[6];
    float* out = (float*)d_out;

    const int B = in_sizes[0] / T_LEN;   // 4096
    const int grid = B / 16;             // 256 independent waves
    lstm_wmma_kernel<<<grid, 32, 0, stream>>>(x, emb, Wx, Wh, bg, W_fc, b_fc, out);
}
